// InfoNCE_25958782337420
// MI455X (gfx1250) — compile-verified, matrix operand order checked by reading, not run on hardware
//
#include <hip/hip_runtime.h>
#include <math.h>

#define NS 512   // N samples
#define XD 128   // X_DIM == Y_DIM
#define HD 512   // hidden H

typedef __attribute__((ext_vector_type(16))) __bf16 v16bf;
typedef __attribute__((ext_vector_type(8)))  __bf16 v8bf;
typedef __attribute__((ext_vector_type(8)))  float  v8f;
typedef __attribute__((ext_vector_type(4)))  int    v4i;

union BF16x16 { v16bf v; v8bf h[2]; };

typedef __attribute__((address_space(1))) v4i* as1_v4i;   // global (__device__)
typedef __attribute__((address_space(3))) v4i* as3_v4i;   // LDS (__shared__)

__device__ __forceinline__ float softplusf(float x) {
  return fmaxf(x, 0.f) + log1pf(expf(-fabsf(x)));
}

// 16-byte global -> LDS copy. Prefer the CDNA5 async-to-LDS path (ASYNCcnt).
__device__ __forceinline__ void copy16_g2l(const __bf16* g, __bf16* l) {
#if __has_builtin(__builtin_amdgcn_global_load_async_to_lds_b128)
  __builtin_amdgcn_global_load_async_to_lds_b128(
      (as1_v4i)(unsigned long long)g,
      (as3_v4i)(unsigned int)(unsigned long long)l, 0, 0);
#elif defined(__AMDGCN__)
  unsigned int loff = (unsigned int)(unsigned long long)l;   // LDS byte offset
  asm volatile("global_load_async_to_lds_b128 %0, %1, off"
               :: "v"(loff), "v"((unsigned long long)g) : "memory");
#else
  *(v8bf*)l = *(const v8bf*)g;   // host pass placeholder
#endif
}

__device__ __forceinline__ void wait_async0() {
#if __has_builtin(__builtin_amdgcn_s_wait_asynccnt)
  __builtin_amdgcn_s_wait_asynccnt(0);
#elif defined(__AMDGCN__)
  asm volatile("s_wait_asynccnt 0" ::: "memory");
#endif
}

// ---------------------------------------------------------------------------
// hx = X @ W1  (tiny: VALU is fine)
// ---------------------------------------------------------------------------
__global__ void __launch_bounds__(HD)
k_h1(const float* __restrict__ X, const float* __restrict__ W,
     float* __restrict__ Hout, int K) {
  const int i = blockIdx.x;
  const int h = threadIdx.x;
  const float* xr = X + i * K;
  float acc = 0.f;
  for (int k = 0; k < K; ++k) acc = fmaf(xr[k], W[k * HD + h], acc);
  Hout[i * HD + h] = acc;
}

// ---------------------------------------------------------------------------
// W2t[n][k] = bf16(W2[k][n])  — column-major bf16 W2 (K contiguous per column)
// ---------------------------------------------------------------------------
__global__ void __launch_bounds__(256)
k_w2t(const float* __restrict__ W2, __bf16* __restrict__ W2t) {
  const int idx = blockIdx.x * 256 + threadIdx.x;
  const int k = idx >> 9;
  const int n = idx & (HD - 1);
  W2t[n * HD + k] = (__bf16)W2[idx];
}

// ---------------------------------------------------------------------------
// Fused pair tail, block = 16 i  x  8 j  (M = 128 pairs), 256 thr = 8 waves.
// Wave w owns j = jbase + w: its A fragment (16 rows x K=512 bf16) lives in
// 128 VGPRs, built once in exact WMMA A lane order.  B (W2t) is staged per
// 64-column chunk into LDS (64 KB) pre-swizzled into exact WMMA B fragment
// order, shared by all 8 waves, copied with global_load_async_to_lds_b128.
// Steady state per WMMA: 2 ds_load_b128 with k-linear addresses, no swizzle
// math.  Each wave covers all N, so it writes FINAL T1 = softplus(. + b3).
// ---------------------------------------------------------------------------
__global__ void __launch_bounds__(256, 1)
k_pair_tail(const float* __restrict__ hx, const float* __restrict__ hy,
            const float* __restrict__ b1, const __bf16* __restrict__ W2t,
            const float* __restrict__ b2, const float* __restrict__ W3,
            const float* __restrict__ b3, float* __restrict__ T1) {
  // B fragment store: block (t,ks) of 512 elems at ((t*16+ks)<<9),
  // lane's two 16B reads are [lane*16] and [+512B] -> fully bank-spread.
  __shared__ __attribute__((aligned(16))) __bf16 Blds[4 * 16 * 512];  // 64 KB

  const int tid   = threadIdx.x;
  const int lane  = tid & 31;
  const int wave  = tid >> 5;
  const int ibase = blockIdx.x * 16;   // 32 i-tiles
  const int jbase = blockIdx.y * 8;    // 64 j-octets
  const int koff  = (lane >> 4) << 3;  // lanes 16-31 hold the K+8 sub-block

  // ---- build register-resident A fragment: rows ii = lane&15, j = jbase+wave
  const float* hxr = hx + (jbase + wave) * HD;          // uniform per wave
  const float* hyr = hy + (ibase + (lane & 15)) * HD;   // per-lane row
  BF16x16 afrag[16];
#pragma unroll
  for (int ks = 0; ks < 16; ++ks) {
    const int kb = ks * 32 + koff;     // elems 0-7: K=kb.. ; 8-15: K=kb+16..
    v8f f0 = *(const v8f*)(hxr + kb)      + *(const v8f*)(hyr + kb)
           + *(const v8f*)(b1 + kb);
    v8f f1 = *(const v8f*)(hxr + kb + 16) + *(const v8f*)(hyr + kb + 16)
           + *(const v8f*)(b1 + kb + 16);
#pragma unroll
    for (int e = 0; e < 8; ++e) {
      f0[e] = fmaxf(f0[e], 0.f);
      f1[e] = fmaxf(f1[e], 0.f);
    }
    afrag[ks].h[0] = __builtin_convertvector(f0, v8bf);
    afrag[ks].h[1] = __builtin_convertvector(f1, v8bf);
  }

  float tac[8];
#pragma unroll
  for (int r = 0; r < 8; ++r) tac[r] = 0.f;

  for (int nc = 0; nc < 8; ++nc) {               // 64 output columns per chunk
    if (nc) __syncthreads();                     // Blds reuse protection

    // ---- stage B chunk: cols [nc*64, nc*64+64) x K=512 into fragment order.
    // 16B source chunk (n=nloc, K=ks*32+cc*8 ..+7) -> 16B LDS chunk.
    const __bf16* Wsrc = W2t + (size_t)nc * (64 * HD);
#pragma unroll 4
    for (int it = 0; it < 16; ++it) {
      const int idx  = tid + (it << 8);          // 0..4095
      const int nloc = idx & 63;
      const int ks   = (idx >> 6) & 15;
      const int cc   = idx >> 10;                // 0..3 (K sub-chunk)
      const __bf16* src = Wsrc + nloc * HD + ks * 32 + cc * 8;
      const int t      = nloc >> 4;
      const int lane_b = ((cc & 1) << 4) | (nloc & 15);
      const int chunk  = (cc >> 1) & 1;
      copy16_g2l(src, &Blds[((t * 16 + ks) << 9) + (chunk << 8) + (lane_b << 3)]);
    }
    wait_async0();
    __syncthreads();

    // ---- 4 N-tiles x 16 K-steps of WMMA; A from registers, B from LDS
    v8f cq[4];
#pragma unroll
    for (int q = 0; q < 4; ++q) cq[q] = (v8f){0.f,0.f,0.f,0.f,0.f,0.f,0.f,0.f};
#pragma unroll
    for (int ks = 0; ks < 16; ++ks) {
#pragma unroll
      for (int q = 0; q < 4; ++q) {
        BF16x16 b;
        const __bf16* bp = &Blds[((q * 16 + ks) << 9) + (lane << 3)];
        b.h[0] = *(const v8bf*)bp;
        b.h[1] = *(const v8bf*)(bp + 256);
        cq[q] = __builtin_amdgcn_wmma_f32_16x16x32_bf16(
                    false, afrag[ks].v, false, b.v, (short)0, cq[q], false, false);
      }
    }

    // ---- fused epilogue: tac += relu(h2 + b2[n]) * W3[n]
#pragma unroll
    for (int q = 0; q < 4; ++q) {
      const int n = nc * 64 + q * 16 + (lane & 15);
      const float b2n = b2[n];
      const float w3n = W3[n];
#pragma unroll
      for (int r = 0; r < 8; ++r) {
        float v = cq[q][r] + b2n;
        v = fmaxf(v, 0.f);
        tac[r] = fmaf(v, w3n, tac[r]);
      }
    }
  }

  // ---- reduce over the 16 N-lanes of each half (C: vgpr r = M r / r+8)
#pragma unroll
  for (int r = 0; r < 8; ++r) {
    float t = tac[r];
    t += __shfl_xor(t, 1, 32);
    t += __shfl_xor(t, 2, 32);
    t += __shfl_xor(t, 4, 32);
    t += __shfl_xor(t, 8, 32);
    tac[r] = t;
  }
  if ((lane & 15) == 0) {
    const float bb3 = b3[0];
    const int j  = jbase + wave;
    const int mi = (lane >> 4) << 3;    // lane 0 -> M=r, lane 16 -> M=r+8
#pragma unroll
    for (int r = 0; r < 8; ++r)
      T1[(ibase + r + mi) * NS + j] = softplusf(tac[r] + bb3);
  }
}

// ---------------------------------------------------------------------------
// Per-row stable logsumexp of final T1.
// ---------------------------------------------------------------------------
__global__ void __launch_bounds__(256)
k_lse(const float* __restrict__ T1, float* __restrict__ lse) {
  __shared__ float red[256];
  const int i   = blockIdx.x;
  const int tid = threadIdx.x;
  const float v0 = T1[i * NS + tid];
  const float v1 = T1[i * NS + tid + 256];
  float m = fmaxf(v0, v1);
  red[tid] = m; __syncthreads();
  for (int s = 128; s > 0; s >>= 1) {
    if (tid < s) red[tid] = fmaxf(red[tid], red[tid + s]);
    __syncthreads();
  }
  m = red[0]; __syncthreads();
  red[tid] = expf(v0 - m) + expf(v1 - m);
  __syncthreads();
  for (int s = 128; s > 0; s >>= 1) {
    if (tid < s) red[tid] += red[tid + s];
    __syncthreads();
  }
  if (tid == 0) lse[i] = m + logf(red[0]);
}

// ---------------------------------------------------------------------------
// Final scalar: mean(diag(T1)) - (mean(lse) - log N).   T0 == diag(T1).
// ---------------------------------------------------------------------------
__global__ void __launch_bounds__(NS)
k_final(const float* __restrict__ T1, const float* __restrict__ lse,
        float* __restrict__ out) {
  __shared__ float rd[NS];
  __shared__ float rl[NS];
  const int i = threadIdx.x;
  rd[i] = T1[i * NS + i];
  rl[i] = lse[i];
  __syncthreads();
  for (int s = NS / 2; s > 0; s >>= 1) {
    if (i < s) { rd[i] += rd[i + s]; rl[i] += rl[i + s]; }
    __syncthreads();
  }
  if (i == 0)
    out[0] = rd[0] / (float)NS - (rl[0] / (float)NS - logf((float)NS));
}

// ---------------------------------------------------------------------------
extern "C" void kernel_launch(void* const* d_in, const int* in_sizes, int n_in,
                              void* d_out, int out_size, void* d_ws, size_t ws_size,
                              hipStream_t stream) {
  const float* x   = (const float*)d_in[0];
  const float* y   = (const float*)d_in[1];
  const float* W1x = (const float*)d_in[2];
  const float* W1y = (const float*)d_in[3];
  const float* b1  = (const float*)d_in[4];
  const float* W2  = (const float*)d_in[5];
  const float* b2  = (const float*)d_in[6];
  const float* W3  = (const float*)d_in[7];
  const float* b3  = (const float*)d_in[8];
  float* out = (float*)d_out;

  char*   ws  = (char*)d_ws;
  float*  hx  = (float*)(ws);                            // 1 MB
  float*  hy  = (float*)(ws + (1u << 20));               // 1 MB
  __bf16* W2t = (__bf16*)(ws + (2u << 20));              // 512 KB
  float*  T1  = (float*)(ws + (2u << 20) + (512u << 10));// 1 MB
  float*  lse = (float*)(ws + (3u << 20) + (512u << 10));// 2 KB

  k_h1<<<NS, HD, 0, stream>>>(x, W1x, hx, XD);
  k_h1<<<NS, HD, 0, stream>>>(y, W1y, hy, XD);
  k_w2t<<<(HD * HD) / 256, 256, 0, stream>>>(W2, W2t);
  k_pair_tail<<<dim3(32, 64), 256, 0, stream>>>(hx, hy, b1, W2t, b2, W3, b3, T1);
  k_lse<<<NS, 256, 0, stream>>>(T1, lse);
  k_final<<<1, NS, 0, stream>>>(T1, lse, out);
}
